// ExplorationCoordinator_90958817394875
// MI455X (gfx1250) — matrix operand order repaired and checked by reading, not compiled, not run on hardware
//
#include <hip/hip_runtime.h>
#include <stdint.h>

#define BB 4
#define SS 2048
#define DD 512
#define HH 8
#define HD 64
#define KC 50

typedef __attribute__((ext_vector_type(16))) __bf16 v16bf;
typedef __attribute__((ext_vector_type(8)))  float  v8f;

union Frag { uint32_t u[8]; v16bf v; };

__device__ __forceinline__ uint16_t f2bf(float f) {
  union { float f; uint32_t u; } c; c.f = f;
  uint32_t u = c.u;
  u += 0x7fffu + ((u >> 16) & 1u);          // round-to-nearest-even
  return (uint16_t)(u >> 16);
}

// ---- fragment loaders (ISA 7.12.2 layouts, wave32) -------------------------
// A 16x32 bf16, row-major source (ld elements). base already at [row0][k0].
__device__ __forceinline__ void loadA_bf16(const uint16_t* base, int ld, Frag& f) {
  int lane = threadIdx.x & 31;
  int m = lane & 15, half = lane >> 4;
  const uint16_t* p = base + (size_t)m * ld;
#pragma unroll
  for (int j = 0; j < 8; ++j) {
    int kp = ((j >> 2) << 4) + (half << 3) + ((j & 3) << 1);
    f.u[j] = *(const uint32_t*)(p + kp);
  }
}
// B 32x16: logical B[k][n] = W[n][k] where entity rows (contiguous over the
// reduction dim, ld elements). base already at [col0][k0].
__device__ __forceinline__ void loadB_nt_bf16(const uint16_t* base, int ld, Frag& f) {
  int lane = threadIdx.x & 31;
  int n = lane & 15, half = lane >> 4;
  const uint16_t* p = base + (size_t)n * ld;
#pragma unroll
  for (int j = 0; j < 8; ++j) f.u[j] = *(const uint32_t*)(p + ((j >> 2) << 4) + (half << 3) + ((j & 3) << 1));
}

__device__ __forceinline__ v8f wmma_bf16(const Frag& a, const Frag& b, v8f c) {
  return __builtin_amdgcn_wmma_f32_16x16x32_bf16(false, a.v, false, b.v,
                                                 (short)0, c, false, false);
}

// ---- K0: fp32 -> bf16 ------------------------------------------------------
__global__ void cvt_f32_bf16(const float* in, uint16_t* out, int n) {
  int i = blockIdx.x * blockDim.x + threadIdx.x;
  if (i < n) out[i] = f2bf(in[i]);
}

// ---- K1: qkv = x @ w_qkv.T + b ; scatter to head-major q,k and transposed v
__global__ __launch_bounds__(256) void qkv_gemm(const uint16_t* xb, const uint16_t* wb,
                                                const float* bias, uint16_t* qbuf,
                                                uint16_t* kbuf, uint16_t* vT) {
  int w = threadIdx.x >> 5, lane = threadIdx.x & 31;
  int row0 = blockIdx.x << 4;                       // 16 rows per block
  int col0 = (blockIdx.y << 7) + (w << 4);          // 128 cols/block, 16/wave
  v8f acc = {};
  for (int k0 = 0; k0 < DD; k0 += 32) {
    Frag a, b;
    loadA_bf16(xb + (size_t)row0 * DD + k0, DD, a);
    loadB_nt_bf16(wb + (size_t)col0 * DD + k0, DD, b);
    acc = wmma_bf16(a, b, acc);
  }
  int n = lane & 15, half = lane >> 4;
  int m = col0 + n;
  float bv = bias[m];
  int sect = m / DD, dd = m % DD;
  int h = dd >> 6, e = dd & 63;
#pragma unroll
  for (int j = 0; j < 8; ++j) {
    int gr = row0 + (half << 3) + j;
    int bi = gr >> 11, s = gr & (SS - 1);
    uint16_t val = f2bf(acc[j] + bv);
    if (sect == 0)      qbuf[(((size_t)(bi * HH + h) * SS) + s) * HD + e] = val;
    else if (sect == 1) kbuf[(((size_t)(bi * HH + h) * SS) + s) * HD + e] = val;
    else                vT[(((size_t)(bi * HH + h) * HD) + e) * SS + s] = val;
  }
}

// ---- K2: per-head scores + softmax (two-pass, WMMA rematerialization).
// Emits probabilities in bf16 (exactly what ctx_gemm consumes) and
// accumulates the head-mean into d_out (single writer => deterministic).
__global__ __launch_bounds__(256) void attn_softmax(const uint16_t* qbuf,
                                                    const uint16_t* kbuf,
                                                    uint16_t* P, float* aw, int h) {
  int w = threadIdx.x >> 5, lane = threadIdx.x & 31;
  int lane15 = lane & 15, half = lane >> 4;
  int bi = blockIdx.x >> 7;                 // 128 row-blocks per batch
  int row0 = (blockIdx.x & 127) << 4;
  const uint16_t* qh = qbuf + ((size_t)(bi * HH + h) * SS) * HD;
  const uint16_t* kh = kbuf + ((size_t)(bi * HH + h) * SS) * HD;
  uint16_t* Pb = P  + (size_t)bi * SS * SS;
  float*   awb = aw + (size_t)bi * SS * SS;

  Frag aq0, aq1;
  loadA_bf16(qh + (size_t)row0 * HD + 0,  HD, aq0);
  loadA_bf16(qh + (size_t)row0 * HD + 32, HD, aq1);

  const float sc = 0.125f;                  // 1/sqrt(64)
  float mloc[8], lloc[8];
#pragma unroll
  for (int j = 0; j < 8; ++j) { mloc[j] = -1e30f; lloc[j] = 0.f; }

  // pass 1: online (max, sumexp) over this wave's 256 columns
  for (int t = 0; t < 16; ++t) {
    int col0 = (w << 8) + (t << 4);
    Frag b0, b1;
    loadB_nt_bf16(kh + (size_t)col0 * HD + 0,  HD, b0);
    loadB_nt_bf16(kh + (size_t)col0 * HD + 32, HD, b1);
    v8f acc = {};
    acc = wmma_bf16(aq0, b0, acc);
    acc = wmma_bf16(aq1, b1, acc);
#pragma unroll
    for (int j = 0; j < 8; ++j) {
      float s = acc[j] * sc;
      float mn = fmaxf(mloc[j], s);
      lloc[j] = lloc[j] * __expf(mloc[j] - mn) + __expf(s - mn);
      mloc[j] = mn;
    }
  }
  // reduce across the 16 lanes of each half (same rows, different cols)
#pragma unroll
  for (int d = 1; d < 16; d <<= 1) {
#pragma unroll
    for (int j = 0; j < 8; ++j) {
      float mo = __shfl_xor(mloc[j], d, 32);
      float lo = __shfl_xor(lloc[j], d, 32);
      float mn = fmaxf(mloc[j], mo);
      lloc[j] = lloc[j] * __expf(mloc[j] - mn) + lo * __expf(mo - mn);
      mloc[j] = mn;
    }
  }
  __shared__ float smax[8][16], ssum[8][16], fm[16], fli[16];
  if (lane15 == 0) {
#pragma unroll
    for (int j = 0; j < 8; ++j) {
      smax[w][(half << 3) + j] = mloc[j];
      ssum[w][(half << 3) + j] = lloc[j];
    }
  }
  __syncthreads();
  if (threadIdx.x < 16) {
    float m = -1e30f, l = 0.f;
    for (int wv = 0; wv < 8; ++wv) {
      float mo = smax[wv][threadIdx.x], lo = ssum[wv][threadIdx.x];
      float mn = fmaxf(m, mo);
      l = l * __expf(m - mn) + lo * __expf(mo - mn);
      m = mn;
    }
    fm[threadIdx.x]  = m;
    fli[threadIdx.x] = 1.f / l;
  }
  __syncthreads();
  float myM[8], myI[8];
#pragma unroll
  for (int j = 0; j < 8; ++j) {
    int r = (half << 3) + j;
    myM[j] = fm[r]; myI[j] = fli[r];
  }
  // pass 2: recompute scores, emit probabilities + head-mean accumulation
  for (int t = 0; t < 16; ++t) {
    int col0 = (w << 8) + (t << 4);
    Frag b0, b1;
    loadB_nt_bf16(kh + (size_t)col0 * HD + 0,  HD, b0);
    loadB_nt_bf16(kh + (size_t)col0 * HD + 32, HD, b1);
    v8f acc = {};
    acc = wmma_bf16(aq0, b0, acc);
    acc = wmma_bf16(aq1, b1, acc);
    int col = col0 + lane15;
#pragma unroll
    for (int j = 0; j < 8; ++j) {
      int r = row0 + (half << 3) + j;
      float p = __expf(acc[j] * sc - myM[j]) * myI[j];
      size_t idx = (size_t)r * SS + col;
      Pb[idx] = f2bf(p);                    // bf16: exactly what ctx_gemm eats
      if (h == 0) awb[idx] = p * 0.125f;    // first head initializes
      else        awb[idx] += p * 0.125f;   // single writer -> deterministic
    }
  }
}

// ---- K3: ctx[:, h*64:(h+1)*64] = P @ v_h (pure bf16 WMMA pipeline) ---------
__global__ __launch_bounds__(256) void ctx_gemm(const uint16_t* P, const uint16_t* vT,
                                                uint16_t* ctx, int h) {
  int w = threadIdx.x >> 5, lane = threadIdx.x & 31;
  int mt = w >> 2, nt = w & 3;
  int growBlk = blockIdx.x << 5;            // 32 rows per block
  int row0 = growBlk + (mt << 4);
  int bi = growBlk >> 11;
  int srow0 = row0 & (SS - 1);
  int e0 = nt << 4;
  const uint16_t* Pb = P + (size_t)bi * SS * SS + (size_t)srow0 * SS;
  const uint16_t* vb = vT + ((size_t)(bi * HH + h) * HD + e0) * SS;
  v8f acc = {};
  for (int k0 = 0; k0 < SS; k0 += 32) {
    Frag a, b;
    loadA_bf16(Pb + k0, SS, a);
    loadB_nt_bf16(vb + k0, SS, b);
    acc = wmma_bf16(a, b, acc);
  }
  int n = lane & 15, half = lane >> 4;
#pragma unroll
  for (int j = 0; j < 8; ++j) {
    int r = row0 + (half << 3) + j;
    ctx[(size_t)r * DD + h * HD + e0 + n] = f2bf(acc[j]);
  }
}

// ---- K4: attended = ctx @ w_out.T + b_out ----------------------------------
__global__ __launch_bounds__(256) void outproj_gemm(const uint16_t* ctx, const uint16_t* wo,
                                                    const float* bias, float* att) {
  int w = threadIdx.x >> 5, lane = threadIdx.x & 31;
  int row0 = blockIdx.x << 4;
  int col0 = (blockIdx.y << 7) + (w << 4);
  v8f acc = {};
  for (int k0 = 0; k0 < DD; k0 += 32) {
    Frag a, b;
    loadA_bf16(ctx + (size_t)row0 * DD + k0, DD, a);
    loadB_nt_bf16(wo + (size_t)col0 * DD + k0, DD, b);
    acc = wmma_bf16(a, b, acc);
  }
  int n = lane & 15, half = lane >> 4;
  int col = col0 + n;
  float bv = bias[col];
#pragma unroll
  for (int j = 0; j < 8; ++j) {
    int r = row0 + (half << 3) + j;
    att[(size_t)r * DD + col] = acc[j] + bv;
  }
}

// ---- K5: soft cluster assignment -------------------------------------------
__global__ __launch_bounds__(64) void cluster_assign(const float* att, const float* centers,
                                                     const float* scales, const float* cw,
                                                     float* outA, float* outW) {
  __shared__ float xrow[DD];
  __shared__ float red[64];
  int n = blockIdx.x;
  int t = threadIdx.x;
  const float* xr = att + (size_t)n * DD;
  float part = 0.f;
  for (int i = t; i < DD; i += 64) { float v = xr[i]; xrow[i] = v; part += v * v; }
  red[t] = part;
  __syncthreads();
  for (int s = 32; s > 0; s >>= 1) { if (t < s) red[t] += red[t + s]; __syncthreads(); }
  float xx = red[0];
  __syncthreads();

  float val = -1e30f;
  if (t < KC) {
    const float* c = centers + (size_t)t * DD;
    float dot = 0.f, cc = 0.f;
    for (int i = 0; i < DD; ++i) { float cv = c[i]; dot += cv * xrow[i]; cc += cv * cv; }
    float d2 = xx + cc - 2.f * dot;
    float dist = sqrtf(fmaxf(d2, 1e-12f));
    val = -dist / (scales[t] + 1e-8f);
  }
  red[t] = val;
  __syncthreads();
  for (int s = 32; s > 0; s >>= 1) { if (t < s) red[t] = fmaxf(red[t], red[t + s]); __syncthreads(); }
  float mx = red[0];
  __syncthreads();

  float e = (t < KC) ? __expf(val - mx) : 0.f;
  red[t] = e;
  __syncthreads();
  for (int s = 32; s > 0; s >>= 1) { if (t < s) red[t] += red[t + s]; __syncthreads(); }
  float sum = red[0];
  __syncthreads();

  float a = (t < KC) ? e / sum : 0.f;
  float wv = (t < KC) ? a * cw[t] : 0.f;
  red[t] = wv;
  __syncthreads();
  for (int s = 32; s > 0; s >>= 1) { if (t < s) red[t] += red[t + s]; __syncthreads(); }
  float wsum = red[0];
  if (t < KC) {
    outA[(size_t)n * KC + t] = a;
    outW[(size_t)n * KC + t] = wv / (wsum + 1e-8f);
  }
}

extern "C" void kernel_launch(void* const* d_in, const int* in_sizes, int n_in,
                              void* d_out, int out_size, void* d_ws, size_t ws_size,
                              hipStream_t stream) {
  const float* x       = (const float*)d_in[0];
  const float* w_qkv   = (const float*)d_in[1];
  const float* b_qkv   = (const float*)d_in[2];
  const float* w_out   = (const float*)d_in[3];
  const float* b_out   = (const float*)d_in[4];
  const float* centers = (const float*)d_in[5];
  const float* scales  = (const float*)d_in[6];
  const float* cweights= (const float*)d_in[7];

  // workspace layout (~94 MB total)
  char* ws = (char*)d_ws;
  size_t off = 0;
  auto alloc = [&](size_t bytes) {
    char* p = ws + off; off += (bytes + 255) & ~(size_t)255; return p;
  };
  uint16_t* xb    = (uint16_t*)alloc((size_t)BB * SS * DD * 2);
  uint16_t* wqkvb = (uint16_t*)alloc((size_t)3 * DD * DD * 2);
  uint16_t* woutb = (uint16_t*)alloc((size_t)DD * DD * 2);
  uint16_t* qb    = (uint16_t*)alloc((size_t)BB * HH * SS * HD * 2);
  uint16_t* kb    = (uint16_t*)alloc((size_t)BB * HH * SS * HD * 2);
  uint16_t* vTb   = (uint16_t*)alloc((size_t)BB * HH * SS * HD * 2);
  uint16_t* ctxb  = (uint16_t*)alloc((size_t)BB * SS * DD * 2);
  uint16_t* Pws   = (uint16_t*)alloc((size_t)BB * SS * SS * 2);   // bf16 probs
  float*    attd  = (float*)   alloc((size_t)BB * SS * DD * 4);

  float* outA  = (float*)d_out;                       // assignments [B,S,K]
  float* outW  = outA + (size_t)BB * SS * KC;         // wa          [B,S,K]
  float* outAW = outW + (size_t)BB * SS * KC;         // attn mean   [B,S,S]

  int nx = BB * SS * DD;
  cvt_f32_bf16<<<(nx + 255) / 256, 256, 0, stream>>>(x, xb, nx);
  int nw = 3 * DD * DD;
  cvt_f32_bf16<<<(nw + 255) / 256, 256, 0, stream>>>(w_qkv, wqkvb, nw);
  int no = DD * DD;
  cvt_f32_bf16<<<(no + 255) / 256, 256, 0, stream>>>(w_out, woutb, no);

  dim3 g1(BB * SS / 16, 3 * DD / 128);
  qkv_gemm<<<g1, 256, 0, stream>>>(xb, wqkvb, b_qkv, qb, kb, vTb);

  for (int h = 0; h < HH; ++h) {
    attn_softmax<<<BB * SS / 16, 256, 0, stream>>>(qb, kb, Pws, outAW, h);
    ctx_gemm<<<BB * SS / 32, 256, 0, stream>>>(Pws, vTb, ctxb, h);
  }

  dim3 g4(BB * SS / 16, DD / 128);
  outproj_gemm<<<g4, 256, 0, stream>>>(ctxb, woutb, b_out, attd);

  cluster_assign<<<BB * SS, 64, 0, stream>>>(attd, centers, scales, cweights, outA, outW);
}
